// GATGraphNet_58033598103933
// MI455X (gfx1250) — matrix operand order repaired and checked by reading, not compiled, not run on hardware
//
#include <hip/hip_runtime.h>
#include <hip/hip_bf16.h>

// ---------------------------------------------------------------------------
// GATv2 + GraphNet pipeline for MI455X (gfx1250, wave32, WMMA).
// All heavy math goes through a tiled v_wmma_f32_16x16x32_f16 GEMM.
// Weights are pre-transposed to [N, Kpad] (K-major) so both A and B tiles
// stage with b128 loads and both fragment reads are contiguous ds_load_b128.
// ---------------------------------------------------------------------------

typedef __attribute__((ext_vector_type(16))) _Float16 v16h;
typedef __attribute__((ext_vector_type(8)))  float    v8f;

#define BM 128
#define BN 128
#define BK 32

// ---------------------------------------------------------------------------
// WMMA GEMM: C[M,N] = act(A[M,K](f16) @ BT[N,K](f16)^T + bias[N])
//   A  : row-major [M, K]      (lda = K)
//   BT : K-major    [N, K]     (ldb = K)  -- i.e. B transposed
// Requires M % 128 == 0 and K % 32 == 0 (all call sites satisfy this).
// N arbitrary (guarded). Output f16 or f32 per flag.
// Block: 256 threads = 8 waves (2 M-slabs x 4 N-slabs); each wave owns a
// 64x32 tile = 4x2 accumulators of 16x16 (v8f each) -> 8 WMMAs per K-slab.
// ---------------------------------------------------------------------------
__global__ __launch_bounds__(256) void k_gemm_wmma(
    const _Float16* __restrict__ A, const _Float16* __restrict__ BT,
    const float* __restrict__ bias, void* __restrict__ Cout,
    int M, int N, int K, int relu, int f16out)
{
  __shared__ __align__(16) _Float16 As [BM][BK + 8];   // +8 halfs pad, 80B row stride
  __shared__ __align__(16) _Float16 BsT[BN][BK + 8];   // transposed: [n][k]

  const int t    = threadIdx.x;
  const int lane = t & 31;
  const int wave = t >> 5;        // 0..7
  const int wm   = wave >> 2;     // 0..1 -> 64-row slab
  const int wn   = wave & 3;      // 0..3 -> 32-col slab
  const int m0   = blockIdx.y * BM;
  const int n0   = blockIdx.x * BN;

  v8f zacc = {};
  v8f acc[4][2];
#pragma unroll
  for (int i = 0; i < 4; ++i)
#pragma unroll
    for (int j = 0; j < 2; ++j) acc[i][j] = zacc;

  // ISA 7.12.2 wave32 fragment coordinates
  const int mrow = lane & 15;          // A: row within 16x32 tile
  const int ksel = (lane >> 4) * 8;    // A: K phase (0 or 8)
  const int ncol = lane & 15;          // B/D: column within 16
  const int kb   = (lane >> 4) * 16;   // unused: B reads use ksel, same k order as A

  for (int k0 = 0; k0 < K; k0 += BK) {
    // ---- stage tiles into LDS: each thread moves 2 x 16B for A and for BT
#pragma unroll
    for (int p = 0; p < 2; ++p) {
      int idx = p * 256 + t;
      int r   = idx >> 2;              // 0..127
      int cb  = (idx & 3) * 8;         // 0,8,16,24 (K offset)
      uint4 av = *(const uint4*)(A + (size_t)(m0 + r) * K + k0 + cb);
      *(uint4*)&As[r][cb] = av;

      uint4 bv;
      if (n0 + r < N)
        bv = *(const uint4*)(BT + (size_t)(n0 + r) * K + k0 + cb);
      else { bv.x = 0u; bv.y = 0u; bv.z = 0u; bv.w = 0u; }
      *(uint4*)&BsT[r][cb] = bv;
    }
    if (k0 + BK < K) {                 // gfx1250 global_prefetch for next K-slab
      __builtin_prefetch(A  + (size_t)(m0 + (t >> 1)) * K + k0 + BK, 0, 1);
      __builtin_prefetch(BT + (size_t)(n0 + (t >> 1)) * K + k0 + BK, 0, 1);
    }
    __syncthreads();

    // ---- fragments (all reads contiguous 16B) + 8 WMMAs per K-slab
    v16h bfrag[2];
#pragma unroll
    for (int nf = 0; nf < 2; ++nf) {
      int nb = wn * 32 + nf * 16 + ncol;
#pragma unroll
      for (int i = 0; i < 16; ++i) bfrag[nf][i] = BsT[nb][((i & 8) << 1) + ksel + (i & 7)];
    }
#pragma unroll
    for (int mf = 0; mf < 4; ++mf) {
      int mb = wm * 64 + mf * 16 + mrow;
      v16h afrag;
#pragma unroll
      for (int i = 0; i < 8; ++i) {
        afrag[i]     = As[mb][ksel + i];
        afrag[8 + i] = As[mb][16 + ksel + i];
      }
#pragma unroll
      for (int nf = 0; nf < 2; ++nf)
        acc[mf][nf] = __builtin_amdgcn_wmma_f32_16x16x32_f16(
            false, afrag, false, bfrag[nf], (short)0, acc[mf][nf], false, false);
    }
    __syncthreads();
  }

  // ---- epilogue: D layout = lanes 0-15 -> M 0..7, lanes 16-31 -> M 8..15
  const int mhi = (lane >> 4) * 8;
#pragma unroll
  for (int mf = 0; mf < 4; ++mf) {
#pragma unroll
    for (int nf = 0; nf < 2; ++nf) {
      int col = n0 + wn * 32 + nf * 16 + ncol;
      if (col >= N) continue;
      float bv = bias ? bias[col] : 0.0f;
#pragma unroll
      for (int r = 0; r < 8; ++r) {
        int row = m0 + wm * 64 + mf * 16 + mhi + r;
        float v = acc[mf][nf][r] + bv;
        if (relu && v < 0.0f) v = 0.0f;
        if (f16out) ((_Float16*)Cout)[(size_t)row * N + col] = (_Float16)v;
        else        ((float*)Cout)[(size_t)row * N + col] = v;
      }
    }
  }
}

// ---------------------------------------------------------------------------
// Elementwise / utility kernels
// ---------------------------------------------------------------------------
__global__ void k_cast_f16(const float* __restrict__ s, _Float16* __restrict__ d, size_t n) {
  size_t i = blockIdx.x * (size_t)blockDim.x + threadIdx.x;
  if (i < n) d[i] = (_Float16)s[i];
}

// cast + transpose + K-pad: W[K,N] (f32, row-major) -> WT[N,Kpad] (f16, K-major)
__global__ void k_cast_w_t(const float* __restrict__ s, _Float16* __restrict__ d,
                           int K, int Kpad, int N) {
  size_t i = blockIdx.x * (size_t)blockDim.x + threadIdx.x;
  if (i >= (size_t)N * Kpad) return;
  int n = (int)(i / Kpad), k = (int)(i % Kpad);
  d[i] = (k < K) ? (_Float16)s[(size_t)k * N + n] : (_Float16)0.0f;
}

__global__ void k_fill_f32(float* p, size_t n, float v) {
  size_t i = blockIdx.x * (size_t)blockDim.x + threadIdx.x;
  if (i < n) p[i] = v;
}

// ---------------------------------------------------------------------------
// GATv2 edge kernels (edges = E0 real edges then N self-loops)
// ---------------------------------------------------------------------------
__device__ inline void edge_sd(const int* ei, int E0, int e, int& s, int& d) {
  if (e < E0) { s = ei[e]; d = ei[E0 + e]; }
  else        { s = e - E0; d = e - E0; }
}

__device__ inline float atomicMaxFloat(float* addr, float val) {
  unsigned int* ua = (unsigned int*)addr;
  unsigned int old = *ua;
  while (val > __uint_as_float(old)) {
    unsigned int assumed = old;
    old = atomicCAS(ua, assumed, __float_as_uint(val));
    if (old == assumed) break;
  }
  return __uint_as_float(old);
}

// one wave per edge: logit[e] = att . leaky_relu(xl[src] + xr[dst], 0.2)
__global__ void k_edge_logit(const _Float16* __restrict__ xl, const _Float16* __restrict__ xr,
                             const int* __restrict__ ei, int E0, int ET, int C,
                             const float* __restrict__ att, float* __restrict__ logit) {
  int e = blockIdx.x * (blockDim.x >> 5) + (threadIdx.x >> 5);
  if (e >= ET) return;
  int lane = threadIdx.x & 31;
  int s, d; edge_sd(ei, E0, e, s, d);
  const _Float16* pl = xl + (size_t)s * C;
  const _Float16* pr = xr + (size_t)d * C;
  float sum = 0.0f;
  for (int c = lane; c < C; c += 32) {
    float v = (float)pl[c] + (float)pr[c];
    v = (v > 0.0f) ? v : 0.2f * v;
    sum += v * att[c];
  }
  for (int off = 16; off; off >>= 1) sum += __shfl_down(sum, off, 32);
  if (lane == 0) logit[e] = sum;
}

__global__ void k_edge_max(const float* __restrict__ logit, const int* __restrict__ ei,
                           int E0, int ET, float* m) {
  int e = blockIdx.x * blockDim.x + threadIdx.x;
  if (e >= ET) return;
  int s, d; edge_sd(ei, E0, e, s, d);
  atomicMaxFloat(&m[d], logit[e]);
}

__global__ void k_edge_expsum(const float* __restrict__ logit, const int* __restrict__ ei,
                              int E0, int ET, const float* __restrict__ m, float* z) {
  int e = blockIdx.x * blockDim.x + threadIdx.x;
  if (e >= ET) return;
  int s, d; edge_sd(ei, E0, e, s, d);
  atomicAdd(&z[d], expf(logit[e] - m[d]));
}

// one wave per edge: agg[dst] += xl[src] * alpha
__global__ void k_edge_agg(const _Float16* __restrict__ xl, const int* __restrict__ ei,
                           int E0, int ET, int C, const float* __restrict__ logit,
                           const float* __restrict__ m, const float* __restrict__ z,
                           float* __restrict__ agg) {
  int e = blockIdx.x * (blockDim.x >> 5) + (threadIdx.x >> 5);
  if (e >= ET) return;
  int lane = threadIdx.x & 31;
  int s, d; edge_sd(ei, E0, e, s, d);
  float alpha = expf(logit[e] - m[d]) / (z[d] + 1e-16f);
  const _Float16* px = xl + (size_t)s * C;
  float* pa = agg + (size_t)d * C;
  for (int c = lane; c < C; c += 32)
    atomicAdd(&pa[c], (float)px[c] * alpha);
}

// h = relu(agg + gat_bias) cast to f16
__global__ void k_h_final(const float* __restrict__ agg, const float* __restrict__ bias,
                          _Float16* __restrict__ h, size_t n, int C) {
  size_t i = blockIdx.x * (size_t)blockDim.x + threadIdx.x;
  if (i >= n) return;
  float v = agg[i] + bias[i % C];
  h[i] = (_Float16)(v > 0.0f ? v : 0.0f);
}

// ---------------------------------------------------------------------------
// GraphNet concat-builders (rows r = node*48 + head; per-row feature concat)
// ---------------------------------------------------------------------------
__global__ void k_build_in1(const _Float16* __restrict__ h, _Float16* __restrict__ dst, int R) {
  size_t i = blockIdx.x * (size_t)blockDim.x + threadIdx.x;
  if (i >= (size_t)R * 96) return;
  int r = (int)(i / 96), c = (int)(i % 96);
  dst[i] = (c < 64) ? h[(size_t)r * 64 + c] : (_Float16)0.0f;   // u0 == 0, pad == 0
}

__global__ void k_build_in2(const _Float16* __restrict__ nx1, const _Float16* __restrict__ h,
                            const float* __restrict__ nu1, _Float16* __restrict__ dst, int R) {
  size_t i = blockIdx.x * (size_t)blockDim.x + threadIdx.x;
  if (i >= (size_t)R * 192) return;
  int r = (int)(i / 192), c = (int)(i % 192);
  int head = r % 48;
  _Float16 v;
  if      (c < 64)  v = nx1[(size_t)r * 64 + c];
  else if (c < 128) v = h[(size_t)r * 64 + (c - 64)];
  else if (c < 160) v = (_Float16)nu1[head * 32 + (c - 128)];
  else              v = (_Float16)0.0f;                         // u0 zeros + K-pad
  dst[i] = v;
}

__global__ void k_build_fin(const _Float16* __restrict__ nx2, const _Float16* __restrict__ nx1,
                            const _Float16* __restrict__ h, _Float16* __restrict__ dst, int R) {
  size_t i = blockIdx.x * (size_t)blockDim.x + threadIdx.x;
  if (i >= (size_t)R * 192) return;
  int r = (int)(i / 192), c = (int)(i % 192);
  _Float16 v;
  if      (c < 64)  v = nx2[(size_t)r * 64 + c];
  else if (c < 128) v = nx1[(size_t)r * 64 + (c - 64)];
  else              v = h[(size_t)r * 64 + (c - 128)];
  dst[i] = v;
}

// sum nx over nodes -> g[48*64]; grid.y chunks nodes, atomicAdd partials
__global__ void k_reduce_nodes(const _Float16* __restrict__ nx, int Nn, float* g) {
  int idx = blockIdx.x * blockDim.x + threadIdx.x;
  if (idx >= 48 * 64) return;
  int h = idx >> 6, c = idx & 63;
  int nb = blockIdx.y * 256;
  int ne = nb + 256 < Nn ? nb + 256 : Nn;
  float s = 0.0f;
  for (int n = nb; n < ne; ++n) s += (float)nx[((size_t)n * 48 + h) * 64 + c];
  atomicAdd(&g[idx], s);
}

__global__ void k_build_gin(const float* __restrict__ g, float* __restrict__ gin) {
  int i = blockIdx.x * blockDim.x + threadIdx.x;
  if (i >= 48 * 80) return;
  int r = i / 80, c = i % 80;
  gin[i] = (c < 64) ? g[r * 64 + c] : 0.0f;                     // u0 == 0
}

// tiny dense layer (48 rows): out[r,c] = act(in[r,:] . W[:,c] + b[c])
__global__ void k_lin_small(const float* __restrict__ in, const float* __restrict__ W,
                            const float* __restrict__ b, float* __restrict__ out,
                            int R, int Kd, int Nd, int relu) {
  int i = blockIdx.x * blockDim.x + threadIdx.x;
  if (i >= R * Nd) return;
  int r = i / Nd, c = i % Nd;
  float s = b[c];
  for (int k = 0; k < Kd; ++k) s += in[r * Kd + k] * W[k * Nd + c];
  if (relu && s < 0.0f) s = 0.0f;
  out[i] = s;
}

// ---------------------------------------------------------------------------
// Host orchestration
// ---------------------------------------------------------------------------
static inline void gemm(hipStream_t st, const _Float16* A, const _Float16* BT,
                        const float* bias, void* out, int M, int N, int K,
                        int relu, int f16o) {
  dim3 g((N + BN - 1) / BN, M / BM);
  k_gemm_wmma<<<g, 256, 0, st>>>(A, BT, bias, out, M, N, K, relu, f16o);
}

extern "C" void kernel_launch(void* const* d_in, const int* in_sizes, int n_in,
                              void* d_out, int out_size, void* d_ws, size_t ws_size,
                              hipStream_t stream) {
  const int Nn = 4096, C = 3072, E0 = 16384, ET = E0 + Nn, R = Nn * 48;
  (void)in_sizes; (void)n_in; (void)out_size; (void)ws_size;

  // ---- inputs (setup_inputs() flattening order)
  const float* x        = (const float*)d_in[0];
  const int*   ei       = (const int*)d_in[1];
  const float* Wl       = (const float*)d_in[3];
  const float* bl       = (const float*)d_in[4];
  const float* Wr       = (const float*)d_in[5];
  const float* br       = (const float*)d_in[6];
  const float* att      = (const float*)d_in[7];
  const float* gat_bias = (const float*)d_in[8];
  // node_mlps: [9..14] layer0 (W,b)x3 ; [15..20] layer1
  const float* n1W1 = (const float*)d_in[9],  *n1b1 = (const float*)d_in[10];
  const float* n1W2 = (const float*)d_in[11], *n1b2 = (const float*)d_in[12];
  const float* n1W3 = (const float*)d_in[13], *n1b3 = (const float*)d_in[14];
  const float* n2W1 = (const float*)d_in[15], *n2b1 = (const float*)d_in[16];
  const float* n2W2 = (const float*)d_in[17], *n2b2 = (const float*)d_in[18];
  const float* n2W3 = (const float*)d_in[19], *n2b3 = (const float*)d_in[20];
  // global_mlps: [21..26] layer0 ; [27..32] layer1 (unused: feeds only unused u)
  const float* g1W1 = (const float*)d_in[21], *g1b1 = (const float*)d_in[22];
  const float* g1W2 = (const float*)d_in[23], *g1b2 = (const float*)d_in[24];
  const float* g1W3 = (const float*)d_in[25], *g1b3 = (const float*)d_in[26];
  const float* W_out = (const float*)d_in[33], *b_out = (const float*)d_in[34];

  // ---- workspace bump allocator (~660 MB; re-initialized every call)
  char* ws = (char*)d_ws;
  size_t off = 0;
  auto alloc = [&](size_t bytes) -> void* {
    off = (off + 255) & ~(size_t)255;
    void* p = ws + off; off += bytes; return p;
  };
  _Float16* x16   = (_Float16*)alloc((size_t)Nn * C * 2);
  _Float16* WlT16 = (_Float16*)alloc((size_t)C * C * 2);   // [N=C, K=C]
  _Float16* WrT16 = (_Float16*)alloc((size_t)C * C * 2);
  _Float16* xl16  = (_Float16*)alloc((size_t)Nn * C * 2);
  _Float16* xr16  = (_Float16*)alloc((size_t)Nn * C * 2);
  float*    agg   = (float*)alloc((size_t)Nn * C * 4);
  float*    logit = (float*)alloc((size_t)ET * 4);
  float*    mbuf  = (float*)alloc((size_t)Nn * 4);
  float*    zbuf  = (float*)alloc((size_t)Nn * 4);
  _Float16* h1    = (_Float16*)alloc((size_t)Nn * C * 2);
  _Float16* h16   = (_Float16*)alloc((size_t)Nn * C * 2);
  _Float16* nx1   = (_Float16*)alloc((size_t)R * 64 * 2);
  _Float16* nx2   = (_Float16*)alloc((size_t)R * 64 * 2);
  _Float16* in1   = (_Float16*)alloc((size_t)R * 96 * 2);
  _Float16* in2   = (_Float16*)alloc((size_t)R * 192 * 2);
  _Float16* fin   = (_Float16*)alloc((size_t)R * 192 * 2);
  _Float16* hidA  = (_Float16*)alloc((size_t)R * 256 * 2);
  _Float16* hidB  = (_Float16*)alloc((size_t)R * 256 * 2);
  _Float16* wA    = (_Float16*)alloc((size_t)256 * 192 * 2);  // [N=256, K<=192]
  _Float16* wB    = (_Float16*)alloc((size_t)256 * 256 * 2);  // [N=256, K=256]
  _Float16* wC    = (_Float16*)alloc((size_t)64  * 256 * 2);  // [N=64,  K=256]
  _Float16* wO    = (_Float16*)alloc((size_t)64  * 192 * 2);  // [N=64,  K=192]
  float* g1buf = (float*)alloc(48 * 64 * 4);
  float* gin   = (float*)alloc(48 * 80 * 4);
  float* gh1   = (float*)alloc(48 * 256 * 4);
  float* gh2   = (float*)alloc(48 * 256 * 4);
  float* nu1   = (float*)alloc(48 * 32 * 4);

  auto cast = [&](const float* s, _Float16* d, size_t n) {
    k_cast_f16<<<(unsigned)((n + 255) / 256), 256, 0, stream>>>(s, d, n);
  };
  // W[K,N] f32 -> WT[N,Kpad] f16 (transpose + pad)
  auto castWT = [&](const float* s, _Float16* d, int K, int Kpad, int N) {
    size_t n = (size_t)N * Kpad;
    k_cast_w_t<<<(unsigned)((n + 255) / 256), 256, 0, stream>>>(s, d, K, Kpad, N);
  };
  auto fill = [&](float* p, size_t n, float v) {
    k_fill_f32<<<(unsigned)((n + 255) / 256), 256, 0, stream>>>(p, n, v);
  };

  // one GATv2 layer: in16 -> out_h16 = relu(segsoftmax-agg(in@Wl, in@Wr) + bias)
  auto run_gat = [&](const _Float16* in16, _Float16* out_h16) {
    gemm(stream, in16, WlT16, bl, xl16, Nn, C, C, 0, 1);
    gemm(stream, in16, WrT16, br, xr16, Nn, C, C, 0, 1);
    fill(mbuf, Nn, -1e30f);
    fill(zbuf, Nn, 0.0f);
    fill(agg, (size_t)Nn * C, 0.0f);
    k_edge_logit<<<(ET + 7) / 8, 256, 0, stream>>>(xl16, xr16, ei, E0, ET, C, att, logit);
    k_edge_max<<<(ET + 255) / 256, 256, 0, stream>>>(logit, ei, E0, ET, mbuf);
    k_edge_expsum<<<(ET + 255) / 256, 256, 0, stream>>>(logit, ei, E0, ET, mbuf, zbuf);
    k_edge_agg<<<(ET + 7) / 8, 256, 0, stream>>>(xl16, ei, E0, ET, C, logit, mbuf, zbuf, agg);
    k_h_final<<<(unsigned)(((size_t)Nn * C + 255) / 256), 256, 0, stream>>>(
        agg, gat_bias, out_h16, (size_t)Nn * C, C);
  };

  // ---- GAT phase (same weights applied twice, per the reference)
  cast(x, x16, (size_t)Nn * C);
  castWT(Wl, WlT16, C, C, C);
  castWT(Wr, WrT16, C, C, C);
  run_gat(x16, h1);
  run_gat(h1, h16);

  // ---- GraphNet layer 1: node MLP [80(->96)] -> 256 -> 256 -> 64
  k_build_in1<<<(unsigned)(((size_t)R * 96 + 255) / 256), 256, 0, stream>>>(h16, in1, R);
  castWT(n1W1, wA, 80, 96, 256);
  gemm(stream, in1, wA, n1b1, hidA, R, 256, 96, 1, 1);
  castWT(n1W2, wB, 256, 256, 256);
  gemm(stream, hidA, wB, n1b2, hidB, R, 256, 256, 1, 1);
  castWT(n1W3, wC, 256, 256, 64);
  gemm(stream, hidB, wC, n1b3, nx1, R, 64, 256, 0, 1);

  // ---- global MLP 1 (48 rows, scalar) -> nu1[48,32]
  fill(g1buf, 48 * 64, 0.0f);
  k_reduce_nodes<<<dim3(12, (Nn + 255) / 256), 256, 0, stream>>>(nx1, Nn, g1buf);
  k_build_gin<<<(48 * 80 + 255) / 256, 256, 0, stream>>>(g1buf, gin);
  k_lin_small<<<(48 * 256 + 255) / 256, 256, 0, stream>>>(gin, g1W1, g1b1, gh1, 48, 80, 256, 1);
  k_lin_small<<<(48 * 256 + 255) / 256, 256, 0, stream>>>(gh1, g1W2, g1b2, gh2, 48, 256, 256, 1);
  k_lin_small<<<(48 * 32 + 255) / 256, 256, 0, stream>>>(gh2, g1W3, g1b3, nu1, 48, 256, 32, 0);

  // ---- GraphNet layer 2: node MLP [176(->192)] -> 256 -> 256 -> 64
  // (global MLP 2 / agg 2 skipped: their only consumer `u` is unused afterwards)
  k_build_in2<<<(unsigned)(((size_t)R * 192 + 255) / 256), 256, 0, stream>>>(nx1, h16, nu1, in2, R);
  castWT(n2W1, wA, 176, 192, 256);
  gemm(stream, in2, wA, n2b1, hidA, R, 256, 192, 1, 1);
  castWT(n2W2, wB, 256, 256, 256);
  gemm(stream, hidA, wB, n2b2, hidB, R, 256, 256, 1, 1);
  castWT(n2W3, wC, 256, 256, 64);
  gemm(stream, hidB, wC, n2b3, nx2, R, 64, 256, 0, 1);

  // ---- final projection: [nx2 | nx1 | h] (192) @ W_out (192x64) + b_out -> f32 out
  k_build_fin<<<(unsigned)(((size_t)R * 192 + 255) / 256), 256, 0, stream>>>(nx2, nx1, h16, fin, R);
  castWT(W_out, wO, 192, 192, 64);
  gemm(stream, fin, wO, b_out, (float*)d_out, R, 64, 192, 0, 0);
}
